// VersorAttention_23502061043884
// MI455X (gfx1250) — matrix operand order repaired and checked by reading, not compiled, not run on hardware
//
#include <hip/hip_runtime.h>

typedef __attribute__((ext_vector_type(16))) __bf16 v16bf;
typedef __attribute__((ext_vector_type(8)))  __bf16 v8bf;
typedef __attribute__((ext_vector_type(8)))  float  v8f;

// ---------------- compile-time Cl(4,1) tables ----------------
struct Cl41 {
  signed char jidx[32][32];  // j with blade_j * blade_l = +-blade_k
  float sgn[32][32];
  float sig[32];             // scalar part of blade_k * blade_k
};

constexpr int popc5(int x){ int c = 0; for(int i=0;i<6;i++) c += (x>>i)&1; return c; }

constexpr float gpsign(int a, int b){
  int swaps = 0;
  int aa = a >> 1;
  while(aa){ swaps += popc5(aa & b); aa >>= 1; }
  float s = (swaps & 1) ? -1.0f : 1.0f;
  if((a & b) & 16) s = -s;   // e5 (bit 4) squares to -1; e1..e4 to +1
  return s;
}

constexpr Cl41 make_tables(){
  Cl41 t{};
  int masks[32] = {};
  int idx[32]   = {};
  int p = 0;
  for(int g=0; g<=5; g++)
    for(int m=0; m<32; m++)
      if(popc5(m)==g){ masks[p]=m; idx[m]=p; ++p; }
  for(int l=0;l<32;l++)
    for(int k=0;k<32;k++){
      int mj = masks[l] ^ masks[k];
      t.jidx[l][k] = (signed char)idx[mj];
      t.sgn[l][k]  = gpsign(mj, masks[l]);
    }
  for(int k=0;k<32;k++) t.sig[k] = gpsign(masks[k], masks[k]);
  return t;
}

__device__ __constant__ Cl41 g_cl = make_tables();

// ---------------- helpers ----------------
__device__ inline v16bf cat8(v8bf lo, v8bf hi){
  return __builtin_shufflevector(lo, hi, 0,1,2,3,4,5,6,7,8,9,10,11,12,13,14,15);
}
__device__ inline v16bf load_frag16(const __bf16* p0, const __bf16* p1){
  return cat8(*(const v8bf*)p0, *(const v8bf*)p1);
}
__device__ inline v8f wmma_bf16(v16bf a, v16bf b, v8f c){
  return __builtin_amdgcn_wmma_f32_16x16x32_bf16(false, a, false, b, (short)0, c, false, false);
}

// ---------------- weight expansion: BT[n=o*32+k][f=i*32+l] = sgn*w[o,i,j] ----------------
__global__ __launch_bounds__(256) void prep_w(const float* __restrict__ w,
                                              __bf16* __restrict__ bt){
  int tid = blockIdx.x * 256 + threadIdx.x;          // 2048*2048 total
  int n = tid >> 11;
  int f = tid & 2047;
  int o = n >> 5, k = n & 31;
  int i = f >> 5, l = f & 31;
  int j = g_cl.jidx[l][k];
  bt[tid] = (__bf16)(g_cl.sgn[l][k] * w[(o*64 + i)*32 + j]);
}

__global__ __launch_bounds__(256) void f32_to_bf16(const float* __restrict__ in,
                                                   __bf16* __restrict__ out, int n){
  int i = blockIdx.x * 256 + threadIdx.x;
  if(i < n) out[i] = (__bf16)in[i];
}

// ---------------- projection GEMM: C(4096x2048) = A(bf16) @ BT^T, fused epilogue ----
// MODE 0: Q -> normalize, *sig[k]*hd^-0.5, bf16 row-major
// MODE 1: K -> normalize, bf16 row-major
// MODE 2: V -> normalize, bf16 store transposed as VT[(b,h,feat)][s]
// MODE 3: O -> normalize, f32 row-major (final output)
template<int MODE>
__global__ __launch_bounds__(256) void gemm_proj(const __bf16* __restrict__ A,
                                                 const __bf16* __restrict__ BT,
                                                 void* __restrict__ outp){
  const int lane = threadIdx.x & 31;
  const int wave = threadIdx.x >> 5;
  const int l15  = lane & 15;
  const bool lo  = lane < 16;
  const int m0   = blockIdx.y * 32 + (wave & 1) * 16;
  const int nc0  = blockIdx.x * 128 + (wave >> 1) * 32;
  const int ab0  = lo ? 0 : 16;

  const __bf16* arow  = A  + (size_t)(m0 + l15) * 2048;
  const __bf16* b0row = BT + (size_t)(nc0 + l15) * 2048 + (lo ? 0 : 16);
  const __bf16* b1row = b0row + (size_t)16 * 2048;

  v8f acc0 = {0,0,0,0,0,0,0,0};
  v8f acc1 = {0,0,0,0,0,0,0,0};
  for(int kb = 0; kb < 2048; kb += 32){
    __builtin_prefetch(arow  + kb + 256, 0, 0);
    __builtin_prefetch(b0row + kb + 256, 0, 0);
    __builtin_prefetch(b1row + kb + 256, 0, 0);
    v16bf af  = load_frag16(arow + kb + ab0, arow + kb + ab0 + 8);
    v16bf bf0 = load_frag16(b0row + kb, b0row + kb + 8);
    v16bf bf1 = load_frag16(b1row + kb, b1row + kb + 8);
    acc0 = wmma_bf16(af, bf0, acc0);
    acc1 = wmma_bf16(af, bf1, acc1);
  }

  // per-(row, 32-col group) L2 normalization: reduce across 16 lanes of same half
  float sc[8];
  #pragma unroll
  for(int g=0; g<8; g++){
    float s = acc0[g]*acc0[g] + acc1[g]*acc1[g];
    s += __shfl_xor(s, 1);
    s += __shfl_xor(s, 2);
    s += __shfl_xor(s, 4);
    s += __shfl_xor(s, 8);
    sc[g] = rsqrtf(s + 1e-6f);
  }

  const int row0 = m0 + (lo ? 0 : 8);
  const int c0 = nc0 + l15;
  const int c1 = c0 + 16;

  if constexpr (MODE == 0 || MODE == 1){
    __bf16* ob = (__bf16*)outp;
    const float q0m = (MODE==0) ? g_cl.sig[l15]      * 0.35355339059f : 1.0f;
    const float q1m = (MODE==0) ? g_cl.sig[16 + l15] * 0.35355339059f : 1.0f;
    #pragma unroll
    for(int g=0; g<8; g++){
      size_t r = (size_t)(row0 + g);
      ob[r*2048 + c0] = (__bf16)(acc0[g]*sc[g]*q0m);
      ob[r*2048 + c1] = (__bf16)(acc1[g]*sc[g]*q1m);
    }
  } else if constexpr (MODE == 2){
    __bf16* vt = (__bf16*)outp;
    const int bb = (blockIdx.y * 32) >> 11;          // batch
    const int stok = row0 & 2047;                    // token of g==0 element
    const int h  = c0 >> 8;
    const int f0 = c0 & 255;
    const int f1 = c1 & 255;
    v8bf p0, p1;
    #pragma unroll
    for(int g=0; g<8; g++){
      p0[g] = (__bf16)(acc0[g]*sc[g]);
      p1[g] = (__bf16)(acc1[g]*sc[g]);
    }
    *(v8bf*)(vt + ((size_t)(bb*8 + h)*256 + f0)*2048 + stok) = p0;
    *(v8bf*)(vt + ((size_t)(bb*8 + h)*256 + f1)*2048 + stok) = p1;
  } else {
    float* of = (float*)outp;
    #pragma unroll
    for(int g=0; g<8; g++){
      size_t r = (size_t)(row0 + g);
      of[r*2048 + c0] = acc0[g]*sc[g];
      of[r*2048 + c1] = acc1[g]*sc[g];
    }
  }
}

// ---------------- attention: one WG = 32 queries of one (b,h); full 2048-key
// score panel lives in LDS (32*2048 bf16 = 128 KB of the 320 KB WGP LDS) -------------
__global__ __launch_bounds__(256) void attn_kernel(const __bf16* __restrict__ Qb,
                                                   const __bf16* __restrict__ Kb,
                                                   const __bf16* __restrict__ VT,
                                                   __bf16* __restrict__ Ob){
  extern __shared__ __bf16 sS[];             // [32][2048] scores -> probs (in place)
  __shared__ float red[32][8];
  __shared__ float rowsum_s[32];
  __shared__ float rowmax_s[32];

  const int b  = blockIdx.z;
  const int h  = blockIdx.y;
  const int q0 = blockIdx.x * 32;
  const int lane = threadIdx.x & 31;
  const int wave = threadIdx.x >> 5;
  const int l15  = lane & 15;
  const bool lo  = lane < 16;
  const int mblock = wave & 1;
  const int ab0 = lo ? 0 : 16;

  // preload this wave's 16x256 Q tile as 8 A-fragments (sig & 1/sqrt(hd) pre-folded)
  v16bf qf[8];
  {
    const __bf16* qp = Qb + (size_t)(b*2048 + q0 + mblock*16 + l15) * 2048 + h*256;
    #pragma unroll
    for(int kk=0; kk<8; kk++)
      qf[kk] = load_frag16(qp + kk*32 + ab0, qp + kk*32 + ab0 + 8);
  }

  // scores: each wave covers 512 keys for its 16-row block
  {
    const int keyb = (wave >> 1) * 512;
    const __bf16* kbase = Kb + (size_t)b*2048*2048 + (size_t)h*256;
    for(int nt=0; nt<32; nt++){
      const int key0 = keyb + nt*16;
      const __bf16* kp = kbase + (size_t)(key0 + l15)*2048 + (lo ? 0 : 16);
      v8f acc = {0,0,0,0,0,0,0,0};
      #pragma unroll
      for(int kk=0; kk<8; kk++){
        v16bf bf = load_frag16(kp + kk*32, kp + kk*32 + 8);
        acc = wmma_bf16(qf[kk], bf, acc);
      }
      const int col = key0 + l15;
      const int r0  = mblock*16 + (lo ? 0 : 8);
      #pragma unroll
      for(int g=0; g<8; g++)
        sS[(size_t)(r0+g)*2048 + col] = (__bf16)acc[g];
    }
  }
  __syncthreads();

  // softmax in LDS (probabilities kept unnormalized; divide at epilogue)
  {
    const int r   = threadIdx.x >> 3;
    const int seg = threadIdx.x & 7;
    __bf16* srow = sS + (size_t)r*2048 + seg*256;
    float mx = -3.0e38f;
    for(int i=0;i<256;i++) mx = fmaxf(mx, (float)srow[i]);
    red[r][seg] = mx;
    __syncthreads();
    if(seg == 0){
      float m2 = red[r][0];
      #pragma unroll
      for(int i=1;i<8;i++) m2 = fmaxf(m2, red[r][i]);
      rowmax_s[r] = m2;
    }
    __syncthreads();
    const float m3 = rowmax_s[r];
    float sum = 0.f;
    for(int i=0;i<256;i++){
      float e = __expf((float)srow[i] - m3);
      sum += e;
      srow[i] = (__bf16)e;
    }
    red[r][seg] = sum;
    __syncthreads();
    if(seg == 0){
      float s = 0.f;
      #pragma unroll
      for(int i=0;i<8;i++) s += red[r][i];
      rowsum_s[r] = s;
    }
  }
  __syncthreads();

  // O = P @ V : P A-frags straight from LDS, V transposed so B-frags are contiguous
  const int nb = (wave >> 1) * 64;
  const __bf16* vtb = VT + (size_t)(b*8 + h) * 256 * 2048;
  v8f z = {0,0,0,0,0,0,0,0};
  v8f oacc[4] = {z,z,z,z};
  const __bf16* prow = sS + (size_t)(mblock*16 + l15)*2048;
  for(int kb=0; kb<2048; kb+=32){
    v16bf af = load_frag16(prow + kb + ab0, prow + kb + ab0 + 8);
    #pragma unroll
    for(int t=0;t<4;t++){
      const __bf16* vp = vtb + (size_t)(nb + t*16 + l15)*2048 + kb + (lo ? 0 : 16);
      v16bf bf = load_frag16(vp, vp + 8);
      oacc[t] = wmma_bf16(af, bf, oacc[t]);
    }
  }
  const int r0 = mblock*16 + (lo ? 0 : 8);
  #pragma unroll
  for(int g=0; g<8; g++){
    const float inv = 1.0f / rowsum_s[r0 + g];
    const size_t m = (size_t)(b*2048 + q0 + r0 + g);
    #pragma unroll
    for(int t=0;t<4;t++)
      Ob[m*2048 + h*256 + nb + t*16 + l15] = (__bf16)(oacc[t][g]*inv);
  }
}

// ---------------- host ----------------
extern "C" void kernel_launch(void* const* d_in, const int* in_sizes, int n_in,
                              void* d_out, int out_size, void* d_ws, size_t ws_size,
                              hipStream_t stream){
  const float* x  = (const float*)d_in[0];
  const float* wq = (const float*)d_in[1];
  const float* wk = (const float*)d_in[2];
  const float* wv = (const float*)d_in[3];
  const float* wo = (const float*)d_in[4];

  char* ws = (char*)d_ws;
  const size_t MB = 1024u*1024u;
  __bf16* W2q = (__bf16*)(ws +  0*MB);
  __bf16* W2k = (__bf16*)(ws +  8*MB);
  __bf16* W2v = (__bf16*)(ws + 16*MB);
  __bf16* W2o = (__bf16*)(ws + 24*MB);
  __bf16* Xb  = (__bf16*)(ws + 32*MB);
  __bf16* Qb  = (__bf16*)(ws + 48*MB);
  __bf16* Kb  = (__bf16*)(ws + 64*MB);
  __bf16* VT  = (__bf16*)(ws + 80*MB);
  __bf16* Ob  = (__bf16*)(ws + 96*MB);

  dim3 thr(256);
  prep_w<<<dim3((2048*2048)/256), thr, 0, stream>>>(wq, W2q);
  prep_w<<<dim3((2048*2048)/256), thr, 0, stream>>>(wk, W2k);
  prep_w<<<dim3((2048*2048)/256), thr, 0, stream>>>(wv, W2v);
  prep_w<<<dim3((2048*2048)/256), thr, 0, stream>>>(wo, W2o);
  f32_to_bf16<<<dim3((4096*2048)/256), thr, 0, stream>>>(x, Xb, 4096*2048);

  dim3 gg(16, 128);
  gemm_proj<0><<<gg, thr, 0, stream>>>(Xb, W2q, (void*)Qb);
  gemm_proj<1><<<gg, thr, 0, stream>>>(Xb, W2k, (void*)Kb);
  gemm_proj<2><<<gg, thr, 0, stream>>>(Xb, W2v, (void*)VT);

  attn_kernel<<<dim3(64, 8, 2), thr, 32*2048*sizeof(__bf16), stream>>>(Qb, Kb, VT, Ob);

  gemm_proj<3><<<gg, thr, 0, stream>>>(Ob, W2o, d_out);

  (void)in_sizes; (void)n_in; (void)out_size; (void)ws_size;
}
